// PosDecoderLayer_42210938585421
// MI455X (gfx1250) — matrix operand order repaired and checked by reading, not compiled, not run on hardware
//
#include <hip/hip_runtime.h>
#include <hip/hip_bf16.h>
#include <stdint.h>

// ---------------------------------------------------------------------------
// PosDecoderLayer for MI455X (gfx1250, wave32, WMMA).
// T=B=32, D=256, H=8, DH=32, N=4096, FF=1024.
// Cross-attention is algebraically folded (single query per batch) so the
// 134MB encoded tensors are streamed only ~3x total; all dense projections
// use v_wmma_f32_16x16x32_bf16 with fp32 accumulation; self-attention K/V
// tiles are staged with CDNA5 async global->LDS copies.
// ---------------------------------------------------------------------------

#define T_   32
#define B_   32
#define D_   256
#define H_   8
#define N_   4096
#define FF_  1024
#define EPS_ 1e-5f

typedef __bf16        v16bf __attribute__((ext_vector_type(16)));
typedef float         v8f   __attribute__((ext_vector_type(8)));
typedef unsigned int  v8u   __attribute__((ext_vector_type(8)));

__device__ __forceinline__ unsigned short f2bf(float f) {
  unsigned u = __builtin_bit_cast(unsigned, f);
  u += 0x7FFFu + ((u >> 16) & 1u);           // round-to-nearest-even
  return (unsigned short)(u >> 16);
}

// CDNA5 async copy: 16B global -> LDS, tracked by ASYNCcnt (no VGPR data path).
// LDS byte address = low 32 bits of flat shared address (ISA 10.2 aperture).
__device__ __forceinline__ void async_copy_b128(void* lds_ptr, const void* gptr) {
  unsigned lds_off = (unsigned)(uintptr_t)lds_ptr;
  unsigned long long ga = (unsigned long long)(uintptr_t)gptr;
  asm volatile("global_load_async_to_lds_b128 %0, %1, off"
               :: "v"(lds_off), "v"(ga) : "memory");
}
__device__ __forceinline__ void async_wait0() {
  asm volatile("s_wait_asynccnt 0x0" ::: "memory");
}

// ---------------------------------------------------------------------------
// Generic bf16-WMMA GEMM:  C[M,Nout] = act(alpha * (sum_t A_t @ W_t^T + bias))
// W_t row-major [Nout, K]; A_t row-major with row stride lda.
// Block: 256 threads (8 wave32 waves). Tile 128x128, K-tile 64, double-
// buffered LDS (2 x 32KB) so global fp32->bf16 staging of tile g+1 overlaps
// the WMMA consumption of tile g. Tile indices advanced incrementally (no
// runtime div/mod in the hot loop).
// ---------------------------------------------------------------------------
template<int NT, bool RELU>
__global__ __launch_bounds__(256) void gemm_bf16(
    const float* __restrict__ A0, const float* __restrict__ A1,
    const float* __restrict__ A2, int lda,
    const float* __restrict__ W0, const float* __restrict__ W1,
    const float* __restrict__ W2,
    const float* __restrict__ bias,
    float* __restrict__ C, int ldc,
    int M, int K, int Nout, float alpha)
{
  __shared__ unsigned short As[2][128 * 64];   // [m][k] bf16
  __shared__ unsigned short Bs[2][128 * 64];   // [n][k] bf16 (W rows, k contiguous)

  const int tid  = threadIdx.x;
  const int bm0  = blockIdx.x * 128;
  const int bn0  = blockIdx.y * 128;
  const int lane = tid & 31;
  const int wv   = tid >> 5;
  const int wm   = (wv & 3) * 32;           // wave row base in tile
  const int wn   = (wv >> 2) * 64;          // wave col base in tile
  const int lg   = lane >> 4;               // lane group (0/1)
  const int lm   = lane & 15;

  const v8f zero8 = {0.f, 0.f, 0.f, 0.f, 0.f, 0.f, 0.f, 0.f};
  v8f acc[2][4];
#pragma unroll
  for (int mt = 0; mt < 2; ++mt)
#pragma unroll
    for (int nt = 0; nt < 4; ++nt) acc[mt][nt] = zero8;

  const int nkb   = K / 64;
  const int niter = NT * nkb;
  const bool full = (bm0 + 128 <= M) && (bn0 + 128 <= Nout);

  auto stage = [&](const float* __restrict__ A, const float* __restrict__ W,
                   int kb, int bufi) {
    unsigned short* Ad = &As[bufi][0];
    unsigned short* Bd = &Bs[bufi][0];
#pragma unroll
    for (int it = 0; it < 8; ++it) {
      int idx  = tid * 4 + it * 1024;
      int rrow = idx >> 6;
      int col  = idx & 63;
      float4 av, wvv;
      if (full) {
        av  = *(const float4*)(A + (size_t)(bm0 + rrow) * lda + kb * 64 + col);
        wvv = *(const float4*)(W + (size_t)(bn0 + rrow) * K   + kb * 64 + col);
      } else {
        av  = make_float4(0.f, 0.f, 0.f, 0.f);
        wvv = make_float4(0.f, 0.f, 0.f, 0.f);
        if (bm0 + rrow < M)
          av  = *(const float4*)(A + (size_t)(bm0 + rrow) * lda + kb * 64 + col);
        if (bn0 + rrow < Nout)
          wvv = *(const float4*)(W + (size_t)(bn0 + rrow) * K   + kb * 64 + col);
      }
      uint2 pa;
      pa.x = (unsigned)f2bf(av.x) | ((unsigned)f2bf(av.y) << 16);
      pa.y = (unsigned)f2bf(av.z) | ((unsigned)f2bf(av.w) << 16);
      *(uint2*)&Ad[rrow * 64 + col] = pa;
      uint2 pb;
      pb.x = (unsigned)f2bf(wvv.x) | ((unsigned)f2bf(wvv.y) << 16);
      pb.y = (unsigned)f2bf(wvv.z) | ((unsigned)f2bf(wvv.w) << 16);
      *(uint2*)&Bd[rrow * 64 + col] = pb;
      if (full && kb + 1 < nkb) {                 // global_prefetch_b8 next tile
        __builtin_prefetch(A + (size_t)(bm0 + rrow) * lda + (kb + 1) * 64 + col, 0, 1);
        __builtin_prefetch(W + (size_t)(bn0 + rrow) * K   + (kb + 1) * 64 + col, 0, 1);
      }
    }
  };

  auto compute = [&](int bufi) {
    const unsigned short* Ad = &As[bufi][0];
    const unsigned short* Bd = &Bs[bufi][0];
#pragma unroll
    for (int ks = 0; ks < 2; ++ks) {
      const int kbase = ks * 32;
      v8u af[2], bfr[4];
#pragma unroll
      for (int mt = 0; mt < 2; ++mt)
#pragma unroll
        for (int v = 0; v < 8; ++v) {
          // CDNA5 A layout: k = 2*(v&3) + 8*laneGroup + 16*(v>>2)
          int k = kbase + 2 * (v & 3) + 8 * lg + 16 * (v >> 2);
          af[mt][v] = *(const unsigned*)&Ad[(wm + mt * 16 + lm) * 64 + k];
        }
#pragma unroll
      for (int nt = 0; nt < 4; ++nt)
#pragma unroll
        for (int v = 0; v < 8; ++v) {
          // CDNA5 B layout: k = 2*(v&3) + 8*laneGroup + 16*(v>>2)
          int k = kbase + 2 * (v & 3) + 8 * lg + 16 * (v >> 2);
          bfr[nt][v] = *(const unsigned*)&Bd[(wn + nt * 16 + lm) * 64 + k];
        }
#pragma unroll
      for (int mt = 0; mt < 2; ++mt)
#pragma unroll
        for (int nt = 0; nt < 4; ++nt)
          acc[mt][nt] = __builtin_amdgcn_wmma_f32_16x16x32_bf16(
              false, __builtin_bit_cast(v16bf, af[mt]),
              false, __builtin_bit_cast(v16bf, bfr[nt]),
              (short)0, acc[mt][nt], false, false);
    }
  };

  // prologue: stage tile (t=0, kb=0); then advance (tn,kbn) incrementally.
  stage(A0, W0, 0, 0);
  const float* An = A0;
  const float* Wn = W0;
  int tn = 0, kbn = 1;
  if (kbn == nkb) {
    kbn = 0; tn = 1;
    An = (NT > 1) ? A1 : A0; Wn = (NT > 1) ? W1 : W0;
  }
  for (int g = 0; g < niter; ++g) {
    __syncthreads();                        // staging of buffer g&1 done
    if (g + 1 < niter) {
      stage(An, Wn, kbn, (g + 1) & 1);      // overlaps with WMMA below
      if (++kbn == nkb) {                   // advance without div/mod
        kbn = 0; ++tn;
        An = (tn == 1) ? A1 : A2;
        Wn = (tn == 1) ? W1 : W2;
      }
    }
    compute(g & 1);
  }

  // ---- epilogue: C/D layout m = v + 8*laneGroup, n = lane&15 ----
#pragma unroll
  for (int mt = 0; mt < 2; ++mt) {
#pragma unroll
    for (int nt = 0; nt < 4; ++nt) {
      int col = bn0 + wn + nt * 16 + lm;
#pragma unroll
      for (int v = 0; v < 8; ++v) {
        int row = bm0 + wm + mt * 16 + v + 8 * lg;
        if (row < M && col < Nout) {
          float val = acc[mt][nt][v];
          if (bias) val += bias[col];
          val *= alpha;
          if (RELU) val = fmaxf(val, 0.f);
          C[(size_t)row * ldc + col] = val;
        }
      }
    }
  }
}

// ---------------------------------------------------------------------------
// Self attention: per batch b, 8 heads x (32 queries x 32 keys).
// K/V slices staged via CDNA5 async global->LDS copies (ASYNCcnt).
// Also emits weights = attn.mean(over heads) -> [B,T,T].
// ---------------------------------------------------------------------------
__global__ __launch_bounds__(256) void sa_attention(
    const float* __restrict__ qh, const float* __restrict__ kh,
    const float* __restrict__ vh,
    float* __restrict__ attn_out, float* __restrict__ weights)
{
  __shared__ float buf[32 * 256];
  __shared__ float wsum[32 * 32];
  const int b = blockIdx.x, tid = threadIdx.x;
  const int t = tid & 31, h = tid >> 5;
  for (int i = tid * 4; i < 32 * 256; i += 1024) {
    int s = i >> 8, d = i & 255;
    async_copy_b128(&buf[i], &kh[((size_t)s * B_ + b) * D_ + d]);
  }
  for (int i = tid; i < 1024; i += 256) wsum[i] = 0.f;
  async_wait0();
  __syncthreads();

  float qv[32];
#pragma unroll
  for (int d = 0; d < 32; ++d) qv[d] = qh[((size_t)t * B_ + b) * D_ + h * 32 + d];
  float lg[32];
  float mx = -1e30f;
  for (int s = 0; s < 32; ++s) {
    float a = 0.f;
#pragma unroll
    for (int d = 0; d < 32; ++d) a += qv[d] * buf[s * 256 + h * 32 + d];
    lg[s] = a; mx = fmaxf(mx, a);
  }
  float sum = 0.f;
  for (int s = 0; s < 32; ++s) { lg[s] = __expf(lg[s] - mx); sum += lg[s]; }
  float inv = 1.f / sum;
  for (int s = 0; s < 32; ++s) {
    lg[s] *= inv;
    atomicAdd(&wsum[t * 32 + s], lg[s] * (1.f / 8.f));
  }
  __syncthreads();
  for (int i = tid; i < 1024; i += 256) weights[(size_t)b * 1024 + i] = wsum[i];
  __syncthreads();
  for (int i = tid * 4; i < 32 * 256; i += 1024) {
    int s = i >> 8, d = i & 255;
    async_copy_b128(&buf[i], &vh[((size_t)s * B_ + b) * D_ + d]);
  }
  async_wait0();
  __syncthreads();
  float out[32];
#pragma unroll
  for (int d = 0; d < 32; ++d) out[d] = 0.f;
  for (int s = 0; s < 32; ++s) {
    float p = lg[s];
#pragma unroll
    for (int d = 0; d < 32; ++d) out[d] += p * buf[s * 256 + h * 32 + d];
  }
#pragma unroll
  for (int d = 0; d < 32; ++d)
    attn_out[((size_t)t * B_ + b) * D_ + h * 32 + d] = out[d];
}

// ---------------------------------------------------------------------------
// out = LayerNorm(x + add). mode 0: add is full [M,256] (or null).
// mode 1: add is [32,256] applied only to rows with b==0 (cross pad).
// Wave-per-row (wave32), shuffle reductions.
// ---------------------------------------------------------------------------
__global__ __launch_bounds__(256) void add_ln(
    const float* __restrict__ x, const float* __restrict__ add,
    const float* __restrict__ g, const float* __restrict__ bta,
    float* __restrict__ out, int mode)
{
  int row  = blockIdx.x * 8 + (threadIdx.x >> 5);
  int lane = threadIdx.x & 31;
  float v[8];
  float s = 0.f;
#pragma unroll
  for (int i = 0; i < 8; ++i) {
    int d = lane + i * 32;
    float val = x[(size_t)row * 256 + d];
    if (mode == 0) {
      if (add) val += add[(size_t)row * 256 + d];
    } else {
      int bb = row & 31, tt = row >> 5;
      if (bb == 0) val += add[tt * 256 + d];
    }
    v[i] = val; s += val;
  }
#pragma unroll
  for (int off = 16; off > 0; off >>= 1) s += __shfl_xor(s, off, 32);
  float mean = s * (1.f / 256.f);
  float vs = 0.f;
#pragma unroll
  for (int i = 0; i < 8; ++i) { float d0 = v[i] - mean; vs += d0 * d0; }
#pragma unroll
  for (int off = 16; off > 0; off >>= 1) vs += __shfl_xor(vs, off, 32);
  float inv = rsqrtf(vs * (1.f / 256.f) + EPS_);
#pragma unroll
  for (int i = 0; i < 8; ++i) {
    int d = lane + i * 32;
    out[(size_t)row * 256 + d] = (v[i] - mean) * inv * g[d] + bta[d];
  }
}

// ---------------------------------------------------------------------------
// Small helpers
// ---------------------------------------------------------------------------
__global__ void vec_sum3(const float* a, const float* b, const float* c, float* o) {
  int i = threadIdx.x; o[i] = a[i] + b[i] + c[i];
}

// C[256,256] = A @ B (both row-major [256,256])
__global__ __launch_bounds__(256) void compose_nn(
    const float* __restrict__ A, const float* __restrict__ Bm, float* __restrict__ C)
{
  int o = blockIdx.x, i = threadIdx.x;
  float acc = 0.f;
  for (int m = 0; m < 256; ++m) acc += A[o * 256 + m] * Bm[m * 256 + i];
  C[o * 256 + i] = acc;
}

// out[o] = W @ (b1 + b2) + badd
__global__ void compose_bias(const float* W, const float* b1, const float* b2,
                             const float* badd, float* out)
{
  int o = threadIdx.x;
  float acc = badd[o];
  for (int m = 0; m < 256; ++m) {
    float bb = b1[m] + (b2 ? b2[m] : 0.f);
    acc += W[o * 256 + m] * bb;
  }
  out[o] = acc;
}

__global__ void init_max(int* mx) { mx[0] = 0; mx[1] = 0; }
__global__ void red_max(const float* __restrict__ kp, int* mx) {
  size_t i = (size_t)blockIdx.x * 256 + threadIdx.x;     // over B*N
  atomicMax(&mx[0], __float_as_int(kp[i * 2]));
  atomicMax(&mx[1], __float_as_int(kp[i * 2 + 1]));
}

// u_f[j,h,:] = sum_{o in head h} qh[j,o] * Wk_f[o,:]  (same for u_p)
// cdot[j,h]  = sum_{o in head h} qh[j,o] * bk_eff[o]
__global__ __launch_bounds__(256) void compute_u(
    const float* __restrict__ qh, const float* __restrict__ Wkf,
    const float* __restrict__ Wkp, const float* __restrict__ bkeff,
    float* __restrict__ uf, float* __restrict__ up, float* __restrict__ cdot)
{
  int j = blockIdx.x, d = threadIdx.x;
  __shared__ float qs[256];
  qs[d] = qh[j * 256 + d];
  __syncthreads();
  for (int h = 0; h < 8; ++h) {
    float s1 = 0.f, s2 = 0.f;
    for (int oo = 0; oo < 32; ++oo) {
      int o = h * 32 + oo; float qv = qs[o];
      s1 += qv * Wkf[o * 256 + d];
      s2 += qv * Wkp[o * 256 + d];
    }
    uf[(j * 8 + h) * 256 + d] = s1;
    up[(j * 8 + h) * 256 + d] = s2;
  }
  if (d < 8) {
    float c = 0.f;
    for (int oo = 0; oo < 32; ++oo) { int o = d * 32 + oo; c += qs[o] * bkeff[o]; }
    cdot[j * 8 + d] = c;
  }
}

// logits[j,h,s] = Ef[s,j,:].u_f[j,h] + Ep[s,j,:].u_p[j,h] + cdot + gaussian bias
// streams Ef/Ep exactly once; gaussian bias folded in (diag of gm only).
__global__ __launch_bounds__(256) void cross_logits(
    const float* __restrict__ Ef, const float* __restrict__ Ep,
    const float* __restrict__ uf, const float* __restrict__ up,
    const float* __restrict__ cdot,
    const float* __restrict__ gmap, const float* __restrict__ point_pos,
    const float* __restrict__ key_pos, const int* __restrict__ mxb,
    float* __restrict__ logits)
{
  int j = blockIdx.x, chunk = blockIdx.y;
  int tid = threadIdx.x, lane = tid & 31, w = tid >> 5;
  __shared__ float ufs[8 * 256], ups[8 * 256];
  __shared__ float scl[16], ptv[2], cdl[8];
  for (int i = tid; i < 2048; i += 256) { ufs[i] = uf[j * 2048 + i]; ups[i] = up[j * 2048 + i]; }
  if (tid < 16) scl[tid] = gmap[(j * 32 + j) * 32 + 16 + tid];   // scale pairs (sx,sy)/head
  if (tid < 2) {
    float mem = __int_as_float(mxb[tid]) + 1.f;                  // (w,h) = max+1
    ptv[tid] = point_pos[(j * 32 + j) * 2 + tid] * mem;
  }
  if (tid < 8) cdl[tid] = cdot[j * 8 + tid];
  __syncthreads();
  for (int it = 0; it < 64; ++it) {
    int s = chunk * 512 + it * 8 + w;
    size_t base = ((size_t)s * B_ + j) * 256;
    int d0 = lane * 8;
    float4 e0 = *(const float4*)&Ef[base + d0];
    float4 e1 = *(const float4*)&Ef[base + d0 + 4];
    float4 p0 = *(const float4*)&Ep[base + d0];
    float4 p1 = *(const float4*)&Ep[base + d0 + 4];
    float accs[8];
#pragma unroll
    for (int h = 0; h < 8; ++h) {
      const float* u0 = &ufs[h * 256 + d0];
      const float* u1 = &ups[h * 256 + d0];
      accs[h] = e0.x * u0[0] + e0.y * u0[1] + e0.z * u0[2] + e0.w * u0[3]
              + e1.x * u0[4] + e1.y * u0[5] + e1.z * u0[6] + e1.w * u0[7]
              + p0.x * u1[0] + p0.y * u1[1] + p0.z * u1[2] + p0.w * u1[3]
              + p1.x * u1[4] + p1.y * u1[5] + p1.z * u1[6] + p1.w * u1[7];
    }
#pragma unroll
    for (int h = 0; h < 8; ++h)
#pragma unroll
      for (int off = 16; off > 0; off >>= 1) accs[h] += __shfl_xor(accs[h], off, 32);
    if (lane == 0) {
      float kx = key_pos[((size_t)j * N_ + s) * 2]     + 0.5f - ptv[0];
      float ky = key_pos[((size_t)j * N_ + s) * 2 + 1] + 0.5f - ptv[1];
#pragma unroll
      for (int h = 0; h < 8; ++h) {
        float sx = scl[2 * h], sy = scl[2 * h + 1];
        float gb = -fabsf(kx * kx * sx * sx + ky * ky * sy * sy) * 0.125f;
        logits[(size_t)(j * 8 + h) * N_ + s] = accs[h] + cdl[h] + gb;
      }
    }
  }
}

__global__ __launch_bounds__(256) void softmax_rows(float* __restrict__ logits) {
  int r = blockIdx.x, tid = threadIdx.x;
  __shared__ float sb[4096];
  __shared__ float red[256];
  float* row = logits + (size_t)r * N_;
  float mx = -1e30f;
  for (int i = tid; i < N_; i += 256) { float v = row[i]; sb[i] = v; mx = fmaxf(mx, v); }
  red[tid] = mx; __syncthreads();
  for (int off = 128; off > 0; off >>= 1) { if (tid < off) red[tid] = fmaxf(red[tid], red[tid + off]); __syncthreads(); }
  mx = red[0]; __syncthreads();
  float s = 0.f;
  for (int i = tid; i < N_; i += 256) { float e = __expf(sb[i] - mx); sb[i] = e; s += e; }
  red[tid] = s; __syncthreads();
  for (int off = 128; off > 0; off >>= 1) { if (tid < off) red[tid] += red[tid + off]; __syncthreads(); }
  float inv = 1.f / red[0];
  for (int i = tid; i < N_; i += 256) row[i] = sb[i] * inv;
}

// r[j,h,:] = sum_s p[j,h,s] * Ef[s,j,:]  (streams Ef once per j)
__global__ __launch_bounds__(256) void cross_reduce(
    const float* __restrict__ Ef, const float* __restrict__ p, float* __restrict__ r)
{
  int j = blockIdx.x, tid = threadIdx.x;
  __shared__ float pch[128 * 8];
  float acc[8];
#pragma unroll
  for (int h = 0; h < 8; ++h) acc[h] = 0.f;
  for (int s0 = 0; s0 < N_; s0 += 128) {
    __syncthreads();
#pragma unroll
    for (int q = 0; q < 4; ++q) {
      int li = tid + q * 256;
      int sl = li >> 3, h = li & 7;
      pch[li] = p[(size_t)(j * 8 + h) * N_ + s0 + sl];
    }
    __syncthreads();
    for (int sl = 0; sl < 128; ++sl) {
      float ef = Ef[((size_t)(s0 + sl) * B_ + j) * 256 + tid];
#pragma unroll
      for (int h = 0; h < 8; ++h) acc[h] += pch[sl * 8 + h] * ef;
    }
  }
#pragma unroll
  for (int h = 0; h < 8; ++h) r[(size_t)(j * 8 + h) * 256 + tid] = acc[h];
}

// outc[j,o] = r[j, head(o), :] . Wv_eff[o,:] + bv_eff[o]
__global__ __launch_bounds__(256) void cross_finish(
    const float* __restrict__ r, const float* __restrict__ Wv,
    const float* __restrict__ bv, float* __restrict__ outc)
{
  int j = blockIdx.x, o = threadIdx.x;
  int h = o >> 5;
  const float* rr = r + (size_t)(j * 8 + h) * 256;
  const float* wr = Wv + (size_t)o * 256;
  float acc = bv[o];
  for (int d = 0; d < 256; ++d) acc += rr[d] * wr[d];
  outc[j * 256 + o] = acc;
}

// ---------------------------------------------------------------------------
// Host side
// ---------------------------------------------------------------------------
extern "C" void kernel_launch(void* const* d_in, const int* in_sizes, int n_in,
                              void* d_out, int out_size, void* d_ws, size_t ws_size,
                              hipStream_t stream)
{
  (void)in_sizes; (void)n_in; (void)out_size; (void)ws_size;

  // inputs (setup_inputs dict order; params flattened in insertion order)
  const float* query_tgt  = (const float*)d_in[0];
  const float* query_pos  = (const float*)d_in[1];
  const float* query_time = (const float*)d_in[2];
  const float* query_sine = (const float*)d_in[3];
  const float* Ef         = (const float*)d_in[4];
  const float* Ep         = (const float*)d_in[5];
  const float* key_pos    = (const float*)d_in[6];
  const float* point_pos  = (const float*)d_in[7];
  const float* sa_qc_w = (const float*)d_in[8];  const float* sa_qc_b = (const float*)d_in[9];
  const float* sa_qt_w = (const float*)d_in[10]; const float* sa_qt_b = (const float*)d_in[11];
  const float* sa_qp_w = (const float*)d_in[12]; const float* sa_qp_b = (const float*)d_in[13];
  const float* sa_kc_w = (const float*)d_in[14]; const float* sa_kc_b = (const float*)d_in[15];
  const float* sa_kt_w = (const float*)d_in[16]; const float* sa_kt_b = (const float*)d_in[17];
  const float* sa_kp_w = (const float*)d_in[18]; const float* sa_kp_b = (const float*)d_in[19];
  const float* sa_v_w  = (const float*)d_in[20]; const float* sa_v_b  = (const float*)d_in[21];
  const float* ca_qc_w = (const float*)d_in[22]; const float* ca_qc_b = (const float*)d_in[23];
  const float* ca_kc_w = (const float*)d_in[24]; const float* ca_kc_b = (const float*)d_in[25];
  const float* ca_kp_w = (const float*)d_in[26]; const float* ca_kp_b = (const float*)d_in[27];
  const float* ca_qt_w = (const float*)d_in[28]; const float* ca_qt_b = (const float*)d_in[29];
  const float* ca_v_w  = (const float*)d_in[30]; const float* ca_v_b  = (const float*)d_in[31];
  const float* ca_sine_w = (const float*)d_in[32]; const float* ca_sine_b = (const float*)d_in[33];
  const float* sa_wq  = (const float*)d_in[34]; const float* sa_wk = (const float*)d_in[35];
  const float* sa_wv  = (const float*)d_in[36]; const float* sa_wo = (const float*)d_in[37];
  const float* sa_inb = (const float*)d_in[38]; const float* sa_ob = (const float*)d_in[39];
  const float* ca_wq  = (const float*)d_in[40]; const float* ca_wk = (const float*)d_in[41];
  const float* ca_wv  = (const float*)d_in[42]; const float* ca_wo = (const float*)d_in[43];
  const float* ca_inb = (const float*)d_in[44]; const float* ca_ob = (const float*)d_in[45];
  const float* mlp0_w = (const float*)d_in[46]; const float* mlp0_b = (const float*)d_in[47];
  const float* mlp1_w = (const float*)d_in[48]; const float* mlp1_b = (const float*)d_in[49];
  const float* mlp2_w = (const float*)d_in[50]; const float* mlp2_b = (const float*)d_in[51];
  const float* ffn1_w = (const float*)d_in[52]; const float* ffn1_b = (const float*)d_in[53];
  const float* ffn2_w = (const float*)d_in[54]; const float* ffn2_b = (const float*)d_in[55];
  const float* n1_g = (const float*)d_in[56]; const float* n1_b = (const float*)d_in[57];
  const float* n3_g = (const float*)d_in[58]; const float* n3_b = (const float*)d_in[59];
  const float* n4_g = (const float*)d_in[60]; const float* n4_b = (const float*)d_in[61];

  float* out_tgt = (float*)d_out;                 // [T,B,D]
  float* out_w   = (float*)d_out + T_ * B_ * D_;  // [B,T,T]

  // workspace carve-out
  float* ws = (float*)d_ws;
  size_t off = 0;
  auto alloc = [&](size_t n) { float* p = ws + off; off += n; return p; };
  const size_t MD = (size_t)T_ * B_ * D_;         // 262144
  float* q_sa     = alloc(MD);
  float* k_sa     = alloc(MD);
  float* v_sa     = alloc(MD);
  float* qh       = alloc(MD);
  float* kh       = alloc(MD);
  float* vh       = alloc(MD);
  float* attn_out = alloc(MD);
  float* proj     = alloc(MD);
  float* tgt1     = alloc(MD);
  float* h1       = alloc(MD);
  float* h2       = alloc(MD);
  float* gmap     = alloc((size_t)T_ * B_ * 32);
  float* logits   = alloc((size_t)B_ * H_ * N_);
  float* Wk_f     = alloc(65536);
  float* Wk_p     = alloc(65536);
  float* Wv_eff   = alloc(65536);
  float* bk_eff   = alloc(256);
  float* bv_eff   = alloc(256);
  float* bsum_q   = alloc(256);
  float* bsum_k   = alloc(256);
  float* bsum_qc  = alloc(256);
  float* qc_cross = alloc(8192);
  float* qh_cross = alloc(8192);
  float* crossout = alloc(8192);
  float* tgt2c    = alloc(8192);
  float* u_f      = alloc(65536);
  float* u_p      = alloc(65536);
  float* cdot     = alloc(256);
  float* r_buf    = alloc(65536);
  float* tgt3     = alloc(MD);
  float* ffh      = alloc((size_t)T_ * B_ * FF_);
  float* ffo      = alloc(MD);
  int*   mxb      = (int*)alloc(8);

  const float SCALE = 0.17677669529663687f;   // 32^-0.5
  auto gemm1 = [&](const float* A, int lda, const float* W, const float* bias,
                   float* C, int ldc, int M, int K, int N, float alpha, bool relu) {
    dim3 g((M + 127) / 128, (N + 127) / 128);
    if (relu)
      gemm_bf16<1, true><<<g, 256, 0, stream>>>(A, nullptr, nullptr, lda, W, nullptr,
                                                nullptr, bias, C, ldc, M, K, N, alpha);
    else
      gemm_bf16<1, false><<<g, 256, 0, stream>>>(A, nullptr, nullptr, lda, W, nullptr,
                                                 nullptr, bias, C, ldc, M, K, N, alpha);
  };
  auto gemm3 = [&](const float* A0, const float* A1, const float* A2, int lda,
                   const float* W0, const float* W1, const float* W2,
                   const float* bias, float* C, int ldc, int M, int K, int N) {
    dim3 g((M + 127) / 128, (N + 127) / 128);
    gemm_bf16<3, false><<<g, 256, 0, stream>>>(A0, A1, A2, lda, W0, W1, W2,
                                               bias, C, ldc, M, K, N, 1.f);
  };

  // ---- self attention ----
  vec_sum3<<<1, 256, 0, stream>>>(sa_qc_b, sa_qt_b, sa_qp_b, bsum_q);
  vec_sum3<<<1, 256, 0, stream>>>(sa_kc_b, sa_kt_b, sa_kp_b, bsum_k);
  vec_sum3<<<1, 256, 0, stream>>>(ca_qc_b, ca_sine_b, ca_qt_b, bsum_qc);
  gemm3(query_tgt, query_time, query_pos, 256, sa_qc_w, sa_qt_w, sa_qp_w,
        bsum_q, q_sa, 256, 1024, 256, 256);
  gemm3(query_tgt, query_time, query_pos, 256, sa_kc_w, sa_kt_w, sa_kp_w,
        bsum_k, k_sa, 256, 1024, 256, 256);
  gemm1(query_tgt, 256, sa_v_w, sa_v_b, v_sa, 256, 1024, 256, 256, 1.f, false);
  gemm1(q_sa, 256, sa_wq, sa_inb + 0,   qh, 256, 1024, 256, 256, SCALE, false);
  gemm1(k_sa, 256, sa_wk, sa_inb + 256, kh, 256, 1024, 256, 256, 1.f, false);
  gemm1(v_sa, 256, sa_wv, sa_inb + 512, vh, 256, 1024, 256, 256, 1.f, false);
  sa_attention<<<32, 256, 0, stream>>>(qh, kh, vh, attn_out, out_w);
  gemm1(attn_out, 256, sa_wo, sa_ob, proj, 256, 1024, 256, 256, 1.f, false);
  add_ln<<<128, 256, 0, stream>>>(query_tgt, proj, n1_g, n1_b, tgt1, 0);

  // ---- gaussian MLP + mem (w,h) ----
  gemm1(tgt1, 256, mlp0_w, mlp0_b, h1, 256, 1024, 256, 256, 1.f, true);
  gemm1(h1,   256, mlp1_w, mlp1_b, h2, 256, 1024, 256, 256, 1.f, true);
  gemm1(h2,   256, mlp2_w, mlp2_b, gmap, 32, 1024, 256, 32, 1.f, false);
  init_max<<<1, 1, 0, stream>>>(mxb);
  red_max<<<(B_ * N_) / 256, 256, 0, stream>>>(key_pos, mxb);

  // ---- cross attention (fully folded: no K/V materialization) ----
  gemm3(tgt1, query_sine, query_time, B_ * D_, ca_qc_w, ca_sine_w, ca_qt_w,
        bsum_qc, qc_cross, 256, 32, 256, 256);
  gemm1(qc_cross, 256, ca_wq, ca_inb + 0, qh_cross, 256, 32, 256, 256, SCALE, false);
  compose_nn<<<256, 256, 0, stream>>>(ca_wk, ca_kc_w, Wk_f);
  compose_nn<<<256, 256, 0, stream>>>(ca_wk, ca_kp_w, Wk_p);
  compose_nn<<<256, 256, 0, stream>>>(ca_wv, ca_v_w, Wv_eff);
  compose_bias<<<1, 256, 0, stream>>>(ca_wk, ca_kc_b, ca_kp_b, ca_inb + 256, bk_eff);
  compose_bias<<<1, 256, 0, stream>>>(ca_wv, ca_v_b, nullptr, ca_inb + 512, bv_eff);
  compute_u<<<32, 256, 0, stream>>>(qh_cross, Wk_f, Wk_p, bk_eff, u_f, u_p, cdot);
  cross_logits<<<dim3(32, 8), 256, 0, stream>>>(Ef, Ep, u_f, u_p, cdot, gmap,
                                                point_pos, key_pos, mxb, logits);
  softmax_rows<<<B_ * H_, 256, 0, stream>>>(logits);
  cross_reduce<<<32, 256, 0, stream>>>(Ef, logits, r_buf);
  cross_finish<<<32, 256, 0, stream>>>(r_buf, Wv_eff, bv_eff, crossout);
  gemm1(crossout, 256, ca_wo, ca_ob, tgt2c, 256, 32, 256, 256, 1.f, false);
  add_ln<<<128, 256, 0, stream>>>(tgt1, tgt2c, n3_g, n3_b, tgt3, 1);

  // ---- FFN ----
  gemm1(tgt3, 256, ffn1_w, ffn1_b, ffh, FF_, 1024, 256, FF_, 1.f, true);
  gemm1(ffh, FF_, ffn2_w, ffn2_b, ffo, 256, 1024, FF_, 256, 1.f, false);
  add_ln<<<128, 256, 0, stream>>>(tgt3, ffo, n4_g, n4_b, out_tgt, 0);
}
